// RNN_small_kan_70712341561502
// MI455X (gfx1250) — matrix operand order repaired and checked by reading, not compiled, not run on hardware
//
#include <hip/hip_runtime.h>
#include <math.h>

typedef _Float16 h8   __attribute__((ext_vector_type(8)));
typedef _Float16 v16h __attribute__((ext_vector_type(16)));
typedef float    v8f  __attribute__((ext_vector_type(8)));

#define TSEQ 256
#define LDH  72   // padded LDS row stride in halves (144B, 16B-aligned, conflict-friendly)

// Branch-free tanh: use gfx1250 hardware V_TANH_F32 when available.
__device__ __forceinline__ float fast_tanh(float x) {
#if __has_builtin(__builtin_amdgcn_tanhf)
    return __builtin_amdgcn_tanhf(x);
#else
    // tanh(x) = sign(x) * (1 - 2e/(1+e)), e = exp(-2|x|); branch-free
    float ax = __builtin_fabsf(x);
    float e  = __expf(-2.0f * ax);
    float t  = 1.0f - 2.0f * e / (1.0f + e);
    return __builtin_copysignf(t, x);
#endif
}

// Load one B-fragment (32x16 f16) for WMMA: B[k][c] = W[nc+c][32*kf + k]
// lane: c = lane&15, hi = lane>>4; halves j -> k = 16*hi + j  => 16 consecutive f32 of row (nc+c)
__device__ __forceinline__ v16h load_wfrag(const float* __restrict__ W, int wrow, int kf, int hi) {
    const float* p = W + wrow * 64 + kf * 32 + hi * 16;
    v16h r;
#pragma unroll
    for (int j = 0; j < 16; ++j) r[j] = (_Float16)p[j];
    return r;
}

// Load one A-fragment (16x32 f16) from an LDS tile stored row-major [16][LDH]:
// lane: row = lane&15, hi = lane>>4; halves j (0..15) -> K = 32*kf + 16*hi + j
__device__ __forceinline__ v16h load_afrag(const _Float16* buf, int r, int hi, int kf) {
    const h8* p0 = (const h8*)(buf + r * LDH + kf * 32 + hi * 16);
    const h8* p1 = (const h8*)(buf + r * LDH + kf * 32 + hi * 16 + 8);
    h8 lo = *p0;
    h8 hc = *p1;
    return __builtin_shufflevector(lo, hc, 0,1,2,3,4,5,6,7,8,9,10,11,12,13,14,15);
}

__device__ __forceinline__ v8f wmma_f16(v16h a, v16h b, v8f c) {
    return __builtin_amdgcn_wmma_f32_16x16x32_f16(false, a, false, b, (short)0, c, false, false);
}

// Fused 2-layer tanh RNN. Block = 128 threads = 4 waves; block owns 16 batch rows.
// Wave w computes output columns [16w, 16w+16). Writes final h2 (B x 64) to hout.
__global__ __launch_bounds__(128) void rnn2_fused(
    const float* __restrict__ x,
    const float* __restrict__ wih0, const float* __restrict__ whh0,
    const float* __restrict__ bih0, const float* __restrict__ bhh0,
    const float* __restrict__ wih1, const float* __restrict__ whh1,
    const float* __restrict__ bih1, const float* __restrict__ bhh1,
    float* __restrict__ hout)
{
    __shared__ __align__(16) _Float16 xbuf [16 * LDH];
    __shared__ __align__(16) _Float16 h1buf[16 * LDH];
    __shared__ __align__(16) _Float16 h2buf[16 * LDH];

    const int tid  = threadIdx.x;
    const int lane = tid & 31;
    const int wave = tid >> 5;          // n-tile 0..3
    const int hi   = lane >> 4;
    const int l15  = lane & 15;
    const int nc   = wave * 16;         // first output column of this wave
    const int b0   = blockIdx.x * 16;   // first batch row of this block

    // ---- register-resident weight B-fragments (f16) ----
    v16h Bih0[2], Bhh0[2], Bih1[2], Bhh1[2];
#pragma unroll
    for (int kf = 0; kf < 2; ++kf) {
        Bih0[kf] = load_wfrag(wih0, nc + l15, kf, hi);
        Bhh0[kf] = load_wfrag(whh0, nc + l15, kf, hi);
        Bih1[kf] = load_wfrag(wih1, nc + l15, kf, hi);
        Bhh1[kf] = load_wfrag(whh1, nc + l15, kf, hi);
    }
    const float bias1 = bih0[nc + l15] + bhh0[nc + l15];
    const float bias2 = bih1[nc + l15] + bhh1[nc + l15];

    // ---- x staging: thread -> (row, 8-float chunk) of the 16x64 tile ----
    const int xrow = tid >> 3;
    const int xc8  = (tid & 7) * 8;
    const float* xbase = x + ((size_t)(b0 + xrow)) * (size_t)TSEQ * 64 + xc8;

    float xr[8];
#pragma unroll
    for (int i = 0; i < 8; ++i) xr[i] = xbase[i];   // prefetch t = 0

    // zero recurrent A-fragments
    v16h Ah1[2], Ah2[2], Ax[2];
#pragma unroll
    for (int f = 0; f < 2; ++f)
#pragma unroll
        for (int i = 0; i < 16; ++i) { Ah1[f][i] = (_Float16)0.0f; Ah2[f][i] = (_Float16)0.0f; }

    float gv[8];

    for (int t = 0; t < TSEQ; ++t) {
        // stage x(t) into LDS (f32 -> f16)
        {
            h8 v;
#pragma unroll
            for (int i = 0; i < 8; ++i) v[i] = (_Float16)xr[i];
            *(h8*)(xbuf + xrow * LDH + xc8) = v;
        }
        __syncthreads();

#pragma unroll
        for (int kf = 0; kf < 2; ++kf) Ax[kf] = load_afrag(xbuf, l15, hi, kf);

        // prefetch x(t+1) while WMMAs run
        if (t + 1 < TSEQ) {
#pragma unroll
            for (int i = 0; i < 8; ++i) xr[i] = xbase[(size_t)(t + 1) * 64 + i];
        }

        // ---- layer 1: h1 = tanh(x W_ih0^T + h1 W_hh0^T + b) ----
        v8f D;
#pragma unroll
        for (int i = 0; i < 8; ++i) D[i] = 0.0f;
        D = wmma_f16(Ax[0],  Bih0[0], D);
        D = wmma_f16(Ax[1],  Bih0[1], D);
        D = wmma_f16(Ah1[0], Bhh0[0], D);
        D = wmma_f16(Ah1[1], Bhh0[1], D);
#pragma unroll
        for (int v = 0; v < 8; ++v) {
            float hv = fast_tanh(D[v] + bias1);
            h1buf[(v + 8 * hi) * LDH + nc + l15] = (_Float16)hv;   // D-layout -> row-major LDS
        }
        __syncthreads();

#pragma unroll
        for (int kf = 0; kf < 2; ++kf) Ah1[kf] = load_afrag(h1buf, l15, hi, kf);

        // ---- layer 2: h2 = tanh(h1 W_ih1^T + h2 W_hh1^T + b) ----
        v8f E;
#pragma unroll
        for (int i = 0; i < 8; ++i) E[i] = 0.0f;
        E = wmma_f16(Ah1[0], Bih1[0], E);
        E = wmma_f16(Ah1[1], Bih1[1], E);
        E = wmma_f16(Ah2[0], Bhh1[0], E);
        E = wmma_f16(Ah2[1], Bhh1[1], E);
#pragma unroll
        for (int v = 0; v < 8; ++v) {
            gv[v] = fast_tanh(E[v] + bias2);
            h2buf[(v + 8 * hi) * LDH + nc + l15] = (_Float16)gv[v];
        }
        __syncthreads();

#pragma unroll
        for (int kf = 0; kf < 2; ++kf) Ah2[kf] = load_afrag(h2buf, l15, hi, kf);
    }

    // final hidden state (t = TSEQ-1), full f32
#pragma unroll
    for (int v = 0; v < 8; ++v)
        hout[(size_t)(b0 + v + 8 * hi) * 64 + nc + l15] = gv[v];
}

// ---------------- KAN head: (B,64) -> (B,16) -> (B,1) -> sigmoid ----------------
__device__ __forceinline__ void bspline8(float xv, const float* __restrict__ g, float* __restrict__ out) {
    float Bv[11];
#pragma unroll
    for (int j = 0; j < 11; ++j) Bv[j] = (xv >= g[j] && xv < g[j + 1]) ? 1.0f : 0.0f;
#pragma unroll
    for (int d = 1; d <= 3; ++d) {
        const int n = 11 - d;
        for (int j = 0; j < n; ++j) {
            float left  = (xv - g[j]) / (g[j + d] - g[j]) * Bv[j];
            float right = (g[j + d + 1] - xv) / (g[j + d + 1] - g[j + 1]) * Bv[j + 1];
            Bv[j] = left + right;
        }
    }
#pragma unroll
    for (int j = 0; j < 8; ++j) out[j] = Bv[j];
}

__global__ void kan_head(int B, const float* __restrict__ h,
    const float* __restrict__ g0, const float* __restrict__ c0,
    const float* __restrict__ sb0, const float* __restrict__ sp0,
    const float* __restrict__ g1, const float* __restrict__ c1,
    const float* __restrict__ sb1, const float* __restrict__ sp1,
    float* __restrict__ out)
{
    int b = blockIdx.x * blockDim.x + threadIdx.x;
    if (b >= B) return;
    const float* xr = h + (size_t)b * 64;

    float y1[16];
#pragma unroll
    for (int o = 0; o < 16; ++o) y1[o] = 0.0f;

    for (int i = 0; i < 64; ++i) {
        float xv = xr[i];
        float bas[8];
        bspline8(xv, g0 + i * 12, bas);
        float sil = xv / (1.0f + __expf(-xv));
#pragma unroll
        for (int o = 0; o < 16; ++o) {
            const float* cf = c0 + ((size_t)i * 16 + o) * 8;
            float s = 0.0f;
#pragma unroll
            for (int k = 0; k < 8; ++k) s += bas[k] * cf[k];
            y1[o] += sil * sb0[i * 16 + o] + sp0[i * 16 + o] * s;
        }
    }

    float y2 = 0.0f;
#pragma unroll
    for (int i = 0; i < 16; ++i) {
        float xv = y1[i];
        float bas[8];
        bspline8(xv, g1 + i * 12, bas);
        float sil = xv / (1.0f + __expf(-xv));
        const float* cf = c1 + i * 8;
        float s = 0.0f;
#pragma unroll
        for (int k = 0; k < 8; ++k) s += bas[k] * cf[k];
        y2 += sil * sb1[i] + sp1[i] * s;
    }
    out[b] = 1.0f / (1.0f + __expf(-y2));
}

extern "C" void kernel_launch(void* const* d_in, const int* in_sizes, int n_in,
                              void* d_out, int out_size, void* d_ws, size_t ws_size,
                              hipStream_t stream) {
    const float* x    = (const float*)d_in[0];
    const float* wih0 = (const float*)d_in[1];
    const float* whh0 = (const float*)d_in[2];
    const float* bih0 = (const float*)d_in[3];
    const float* bhh0 = (const float*)d_in[4];
    const float* wih1 = (const float*)d_in[5];
    const float* whh1 = (const float*)d_in[6];
    const float* bih1 = (const float*)d_in[7];
    const float* bhh1 = (const float*)d_in[8];
    const float* g0   = (const float*)d_in[9];
    const float* c0   = (const float*)d_in[10];
    const float* sb0  = (const float*)d_in[11];
    const float* sp0  = (const float*)d_in[12];
    const float* g1   = (const float*)d_in[13];
    const float* c1   = (const float*)d_in[14];
    const float* sb1  = (const float*)d_in[15];
    const float* sp1  = (const float*)d_in[16];

    const int B = in_sizes[0] / (TSEQ * 64);      // 2048
    float* hfin = (float*)d_ws;                   // B x 64 f32 scratch

    rnn2_fused<<<B / 16, 128, 0, stream>>>(x, wih0, whh0, bih0, bhh0,
                                           wih1, whh1, bih1, bhh1, hfin);
    kan_head<<<(B + 127) / 128, 128, 0, stream>>>(B, hfin, g0, c0, sb0, sp0,
                                                  g1, c1, sb1, sp1, (float*)d_out);
}
